// RoPE2DMHA_10857677324353
// MI455X (gfx1250) — matrix operand order repaired and checked by reading, not compiled, not run on hardware
//
#include <hip/hip_runtime.h>

typedef __bf16 bf16_t;
typedef __attribute__((ext_vector_type(8)))  bf16_t v8bf;
typedef __attribute__((ext_vector_type(16))) bf16_t v16bf;
typedef __attribute__((ext_vector_type(8)))  float  v8f;

#define D_MODEL 1024
#define N_HEADS 16
#define D_HEAD  64
#define N_TOK   1024
#define N_BATCH 4
#define ROWS_ALL (N_BATCH * N_TOK)   // 4096

// ---------- helpers ----------

static __device__ __forceinline__ unsigned short f2bf(float f) {
  unsigned int u = __float_as_uint(f);
  u += 0x7FFFu + ((u >> 16) & 1u);   // round-to-nearest-even
  return (unsigned short)(u >> 16);
}

static __device__ __forceinline__ v16bf combine16(v8bf lo, v8bf hi) {
  v16bf r;
#pragma unroll
  for (int i = 0; i < 8; ++i) { r[i] = lo[i]; r[i + 8] = hi[i]; }
  return r;
}

// A-fragment (16x32, M x K): lane holds row (lane%16); element e -> k = (e/8)*16 + 8*half + e%8
// => two 16B chunks at k = 8*half and k = 16 + 8*half
static __device__ __forceinline__ v16bf load_afrag(const bf16_t* row, int half) {
  v8bf lo = *reinterpret_cast<const v8bf*>(row + half * 8);
  v8bf hi = *reinterpret_cast<const v8bf*>(row + 16 + half * 8);
  return combine16(lo, hi);
}

// B-fragment (32x16, K x N): lane holds col (lane%16); element e -> k = (e/8)*16 + 8*half + e%8
// => two 16B chunks at k = 8*half and k = 16 + 8*half
static __device__ __forceinline__ v16bf load_bfrag(const bf16_t* col, int half) {
  const bf16_t* p = col + half * 8;
  v8bf lo = *reinterpret_cast<const v8bf*>(p);
  v8bf hi = *reinterpret_cast<const v8bf*>(p + 16);
  return combine16(lo, hi);
}

static __device__ __forceinline__ v8f wmma_bf16(v16bf a, v16bf b, v8f c) {
  return __builtin_amdgcn_wmma_f32_16x16x32_bf16(false, a, false, b, (short)0, c, false, false);
}

// ---------- kernel 0: fp32 -> bf16 convert ----------

__global__ void cvt_bf16(const float* __restrict__ src, unsigned short* __restrict__ dst, int n) {
  int i = blockIdx.x * blockDim.x + threadIdx.x;
  if (i < n) dst[i] = f2bf(src[i]);
}

// ---------- kernel 1: QKV projection GEMM + fused RoPE ----------
// y = x @ W.T ; x bf16 (4096 x 1024), W bf16 (1024 x 1024) row-major (rows = output cols)
// mode 0 -> Q (rope), 1 -> K (rope), 2 -> V (store transposed)

__global__ __launch_bounds__(128) void qkv_gemm(
    const bf16_t* __restrict__ xb,
    const bf16_t* __restrict__ Wqb, const bf16_t* __restrict__ Wkb, const bf16_t* __restrict__ Wvb,
    const float* __restrict__ cosb, const float* __restrict__ sinb,
    unsigned short* __restrict__ Qb, unsigned short* __restrict__ Kb, unsigned short* __restrict__ Vt)
{
  const int mode = blockIdx.z;
  const bf16_t* W = (mode == 0) ? Wqb : ((mode == 1) ? Wkb : Wvb);

  const int wave = threadIdx.x >> 5;
  const int lane = threadIdx.x & 31;
  const int idx  = lane & 15;
  const int half = lane >> 4;
  const int gw   = blockIdx.x * 4 + wave;      // 0..2047
  const int mt   = gw >> 4;                    // 128 tiles of 32 rows
  const int nt   = gw & 15;                    // 16 tiles of 64 cols
  const int m0   = mt * 32;
  const int n0   = nt * 64;

  v8f acc[2][4];
#pragma unroll
  for (int s = 0; s < 2; ++s)
#pragma unroll
    for (int t = 0; t < 4; ++t)
#pragma unroll
      for (int r = 0; r < 8; ++r) acc[s][t][r] = 0.0f;

  const bf16_t* arow0 = xb + (size_t)(m0 + idx) * D_MODEL;
  const bf16_t* arow1 = xb + (size_t)(m0 + 16 + idx) * D_MODEL;
  const bf16_t* brow[4];
#pragma unroll
  for (int t = 0; t < 4; ++t) brow[t] = W + (size_t)(n0 + t * 16 + idx) * D_MODEL;

  for (int kc = 0; kc < D_MODEL; kc += 32) {
    v16bf a0 = load_afrag(arow0 + kc, half);
    v16bf a1 = load_afrag(arow1 + kc, half);
#pragma unroll
    for (int t = 0; t < 4; ++t) {
      v16bf b = load_bfrag(brow[t] + kc, half);
      acc[0][t] = wmma_bf16(a0, b, acc[0][t]);
      acc[1][t] = wmma_bf16(a1, b, acc[1][t]);
    }
  }

  // epilogue: C/D layout -> element (M = r + 8*half, N = idx) per VGPR r
  if (mode < 2) {
    unsigned short* dst = (mode == 0) ? Qb : Kb;
    const int h = nt;
    const bool odd = (idx & 1);
#pragma unroll
    for (int s = 0; s < 2; ++s) {
#pragma unroll
      for (int t = 0; t < 4; ++t) {
        const int dcol = t * 16 + idx;
        const int csi  = dcol >> 1;            // (axis*16 + pair)
#pragma unroll
        for (int r = 0; r < 8; ++r) {
          const int R   = m0 + s * 16 + r + 8 * half;
          const int bi  = R >> 10;
          const int tok = R & (N_TOK - 1);
          float w  = acc[s][t][r];
          float wp = __shfl_xor(w, 1, 32);     // rope partner (adjacent column)
          float c  = cosb[tok * 32 + csi];
          float sn = sinb[tok * 32 + csi];
          float out = odd ? (sn * wp + c * w) : (c * w - sn * wp);
          dst[((size_t)((bi * N_HEADS + h) * N_TOK + tok)) * D_HEAD + dcol] = f2bf(out);
        }
      }
    }
  } else {
    const int h = nt;
#pragma unroll
    for (int s = 0; s < 2; ++s) {
#pragma unroll
      for (int t = 0; t < 4; ++t) {
        const int dcol = t * 16 + idx;
#pragma unroll
        for (int r = 0; r < 8; ++r) {
          const int R   = m0 + s * 16 + r + 8 * half;
          const int bi  = R >> 10;
          const int tok = R & (N_TOK - 1);
          Vt[((size_t)((bi * N_HEADS + h) * D_HEAD + dcol)) * N_TOK + tok] = f2bf(acc[s][t][r]);
        }
      }
    }
  }
}

// ---------- kernel 2: flash attention (one wave = one (b,h,16-query block)) ----------

__global__ __launch_bounds__(128) void attn_kernel(
    const bf16_t* __restrict__ Qb, const bf16_t* __restrict__ Kb,
    const bf16_t* __restrict__ Vt, unsigned short* __restrict__ Ob)
{
  __shared__ unsigned short pl[4][16 * 32];    // 1KB per wave: P tile (16 rows x 32 keys) bf16

  const int wave = threadIdx.x >> 5;
  const int lane = threadIdx.x & 31;
  const int idx  = lane & 15;
  const int half = lane >> 4;
  const int gw   = blockIdx.x * 4 + wave;      // 0..4095
  const int b    = gw >> 10;
  const int rem  = gw & 1023;
  const int h    = rem >> 6;
  const int q0   = (rem & 63) << 4;

  const size_t bh = (size_t)(b * N_HEADS + h);
  const bf16_t* Qbase = Qb + bh * N_TOK * D_HEAD;
  const bf16_t* Kbase = Kb + bh * N_TOK * D_HEAD;
  const bf16_t* Vbase = Vt + bh * D_HEAD * N_TOK;

  const bf16_t* qrow = Qbase + (size_t)(q0 + idx) * D_HEAD;
  v16bf qf0 = load_afrag(qrow, half);          // d = 0..31
  v16bf qf1 = load_afrag(qrow + 32, half);     // d = 32..63

  v8f o[4];
  float mrun[8], lrun[8];
#pragma unroll
  for (int t = 0; t < 4; ++t)
#pragma unroll
    for (int r = 0; r < 8; ++r) o[t][r] = 0.0f;
#pragma unroll
  for (int r = 0; r < 8; ++r) { mrun[r] = -1e30f; lrun[r] = 0.0f; }

  unsigned short* myl = &pl[wave][0];
  const bf16_t* plb = reinterpret_cast<const bf16_t*>(myl);
  const float scale = 0.125f;                  // 1/sqrt(64)

  for (int k0 = 0; k0 < N_TOK; k0 += 32) {
    // ---- scores: two 16x16 tiles over d=64 (two K-chunks each) ----
    v8f s0, s1;
#pragma unroll
    for (int r = 0; r < 8; ++r) { s0[r] = 0.0f; s1[r] = 0.0f; }
    {
      const bf16_t* kp = Kbase + (size_t)(k0 + idx) * D_HEAD;
      s0 = wmma_bf16(qf0, load_bfrag(kp, half), s0);
      s0 = wmma_bf16(qf1, load_bfrag(kp + 32, half), s0);
    }
    {
      const bf16_t* kp = Kbase + (size_t)(k0 + 16 + idx) * D_HEAD;
      s1 = wmma_bf16(qf0, load_bfrag(kp, half), s1);
      s1 = wmma_bf16(qf1, load_bfrag(kp + 32, half), s1);
    }
#pragma unroll
    for (int r = 0; r < 8; ++r) { s0[r] *= scale; s1[r] *= scale; }

    // ---- online softmax stats (rows r + 8*half live in this 16-lane group) ----
    float cmax[8];
#pragma unroll
    for (int r = 0; r < 8; ++r) cmax[r] = fmaxf(s0[r], s1[r]);
#pragma unroll
    for (int mask = 1; mask <= 8; mask <<= 1)
#pragma unroll
      for (int r = 0; r < 8; ++r) cmax[r] = fmaxf(cmax[r], __shfl_xor(cmax[r], mask, 32));

    float alpha[8], rs[8];
#pragma unroll
    for (int r = 0; r < 8; ++r) {
      float mn = fmaxf(mrun[r], cmax[r]);
      alpha[r] = __expf(mrun[r] - mn);
      mrun[r] = mn;
      float p0 = __expf(s0[r] - mn);
      float p1 = __expf(s1[r] - mn);
      s0[r] = p0; s1[r] = p1;
      rs[r] = p0 + p1;
    }
#pragma unroll
    for (int mask = 1; mask <= 8; mask <<= 1)
#pragma unroll
      for (int r = 0; r < 8; ++r) rs[r] += __shfl_xor(rs[r], mask, 32);
#pragma unroll
    for (int r = 0; r < 8; ++r) lrun[r] = alpha[r] * lrun[r] + rs[r];
#pragma unroll
    for (int t = 0; t < 4; ++t)
#pragma unroll
      for (int r = 0; r < 8; ++r) o[t][r] *= alpha[r];

    // ---- P (C-layout) -> LDS row-major 16x32, then reload as A-fragment ----
#pragma unroll
    for (int r = 0; r < 8; ++r) {
      const int m = r + 8 * half;
      myl[m * 32 + idx]      = f2bf(s0[r]);
      myl[m * 32 + 16 + idx] = f2bf(s1[r]);
    }
    asm volatile("s_wait_dscnt 0" ::: "memory");
    v16bf pf = load_afrag(plb + idx * 32, half);

    // ---- O += P(16x32) @ V(32x64) ----
#pragma unroll
    for (int t = 0; t < 4; ++t) {
      const bf16_t* vp = Vbase + (size_t)(t * 16 + idx) * N_TOK + k0;
      o[t] = wmma_bf16(pf, load_bfrag(vp, half), o[t]);
    }
  }

  // ---- normalize + store O as bf16 [b][n][h*64+d] ----
#pragma unroll
  for (int t = 0; t < 4; ++t) {
#pragma unroll
    for (int r = 0; r < 8; ++r) {
      const int m = r + 8 * half;
      const int q = q0 + m;
      float out = o[t][r] / lrun[r];
      Ob[((size_t)b * N_TOK + q) * D_MODEL + h * D_HEAD + t * 16 + idx] = f2bf(out);
    }
  }
}

// ---------- kernel 3: output projection GEMM + bias, fp32 out ----------

__global__ __launch_bounds__(128) void out_gemm(
    const bf16_t* __restrict__ Ob, const bf16_t* __restrict__ Wob,
    const float* __restrict__ bo, float* __restrict__ Y)
{
  const int wave = threadIdx.x >> 5;
  const int lane = threadIdx.x & 31;
  const int idx  = lane & 15;
  const int half = lane >> 4;
  const int gw   = blockIdx.x * 4 + wave;
  const int mt   = gw >> 4;
  const int nt   = gw & 15;
  const int m0   = mt * 32;
  const int n0   = nt * 64;

  v8f acc[2][4];
#pragma unroll
  for (int s = 0; s < 2; ++s)
#pragma unroll
    for (int t = 0; t < 4; ++t)
#pragma unroll
      for (int r = 0; r < 8; ++r) acc[s][t][r] = 0.0f;

  const bf16_t* arow0 = Ob + (size_t)(m0 + idx) * D_MODEL;
  const bf16_t* arow1 = Ob + (size_t)(m0 + 16 + idx) * D_MODEL;
  const bf16_t* brow[4];
#pragma unroll
  for (int t = 0; t < 4; ++t) brow[t] = Wob + (size_t)(n0 + t * 16 + idx) * D_MODEL;

  for (int kc = 0; kc < D_MODEL; kc += 32) {
    v16bf a0 = load_afrag(arow0 + kc, half);
    v16bf a1 = load_afrag(arow1 + kc, half);
#pragma unroll
    for (int t = 0; t < 4; ++t) {
      v16bf b = load_bfrag(brow[t] + kc, half);
      acc[0][t] = wmma_bf16(a0, b, acc[0][t]);
      acc[1][t] = wmma_bf16(a1, b, acc[1][t]);
    }
  }

#pragma unroll
  for (int s = 0; s < 2; ++s) {
#pragma unroll
    for (int t = 0; t < 4; ++t) {
      const int j = n0 + t * 16 + idx;
      const float bias = bo[j];
#pragma unroll
      for (int r = 0; r < 8; ++r) {
        const int R = m0 + s * 16 + r + 8 * half;
        Y[(size_t)R * D_MODEL + j] = acc[s][t][r] + bias;
      }
    }
  }
}

// ---------- host launch ----------

extern "C" void kernel_launch(void* const* d_in, const int* in_sizes, int n_in,
                              void* d_out, int out_size, void* d_ws, size_t ws_size,
                              hipStream_t stream) {
  (void)in_sizes; (void)n_in; (void)out_size; (void)ws_size;

  const float* x    = (const float*)d_in[0];
  const float* cosb = (const float*)d_in[1];
  const float* sinb = (const float*)d_in[2];
  const float* Wq   = (const float*)d_in[3];
  const float* Wk   = (const float*)d_in[4];
  const float* Wv   = (const float*)d_in[5];
  const float* Wo   = (const float*)d_in[6];
  const float* bo   = (const float*)d_in[7];
  float* Y = (float*)d_out;

  char* w = (char*)d_ws;
  const size_t MB = 1024 * 1024;
  unsigned short* xb  = (unsigned short*)(w + 0);        //  8 MB: x bf16 (4096 x 1024)
  unsigned short* wqb = (unsigned short*)(w + 8  * MB);  //  2 MB
  unsigned short* wkb = (unsigned short*)(w + 10 * MB);  //  2 MB
  unsigned short* wvb = (unsigned short*)(w + 12 * MB);  //  2 MB
  unsigned short* wob = (unsigned short*)(w + 14 * MB);  //  2 MB
  unsigned short* Qb  = (unsigned short*)(w + 16 * MB);  //  8 MB: [b][h][n][64]
  unsigned short* Kb  = (unsigned short*)(w + 24 * MB);  //  8 MB: [b][h][n][64]
  unsigned short* Vt  = (unsigned short*)(w + 32 * MB);  //  8 MB: [b][h][64][n]
  unsigned short* Ob  = (unsigned short*)(w + 40 * MB);  //  8 MB: [b][n][1024]

  const int n_x = ROWS_ALL * D_MODEL;      // 4,194,304
  const int n_w = D_MODEL * D_MODEL;       // 1,048,576
  cvt_bf16<<<(n_x + 255) / 256, 256, 0, stream>>>(x,  xb,  n_x);
  cvt_bf16<<<(n_w + 255) / 256, 256, 0, stream>>>(Wq, wqb, n_w);
  cvt_bf16<<<(n_w + 255) / 256, 256, 0, stream>>>(Wk, wkb, n_w);
  cvt_bf16<<<(n_w + 255) / 256, 256, 0, stream>>>(Wv, wvb, n_w);
  cvt_bf16<<<(n_w + 255) / 256, 256, 0, stream>>>(Wo, wob, n_w);

  qkv_gemm<<<dim3(512, 1, 3), 128, 0, stream>>>(
      (const bf16_t*)xb, (const bf16_t*)wqb, (const bf16_t*)wkb, (const bf16_t*)wvb,
      cosb, sinb, Qb, Kb, Vt);

  attn_kernel<<<1024, 128, 0, stream>>>(
      (const bf16_t*)Qb, (const bf16_t*)Kb, (const bf16_t*)Vt, Ob);

  out_gemm<<<512, 128, 0, stream>>>(
      (const bf16_t*)Ob, (const bf16_t*)wob, bo, Y);
}
